// AMS_63273458204887
// MI455X (gfx1250) — matrix operand order repaired and checked by reading, not compiled, not run on hardware
//
#include <hip/hip_runtime.h>
#include <hip/hip_bf16.h>
#include <math.h>

// Problem constants (match reference)
#define BB 32
#define TT 64
#define NN 32
#define DD 128
#define EE 8
#define HH 2

typedef __attribute__((ext_vector_type(16))) _Float16 v16h;
typedef __attribute__((ext_vector_type(8)))  float    v8f;
typedef __attribute__((ext_vector_type(8)))  unsigned v8u;

#define TS 136   // LDS row stride (halves) for 128-wide f16 tiles (272B = 17*16 -> b128-aligned rows)
#define VS 72    // LDS row stride (halves) for 64-wide f16 tiles (144B = 9*16)

union FragH { v8u u; v16h h; };

// ---- A fragment: 16x32 f16, rows m0.., K k0.., from row-major f16 LDS (stride sr halves)
// ISA layout: lane (0-15): M=lane, pair i -> K = 16*(i>>2) + 2*(i&3)  ; lanes 16-31: +8 on K
__device__ inline v16h load_a(const _Float16* p, int sr, int m0, int k0, int lane) {
  FragH f;
  const int M  = m0 + (lane & 15);
  const int hi = lane >> 4;
  const unsigned* row = (const unsigned*)(p + M * sr);   // sr even -> 4B aligned
#pragma unroll
  for (int i = 0; i < 8; ++i) {
    const int K = k0 + 16 * (i >> 2) + 8 * hi + 2 * (i & 3);
    f.u[i] = row[K >> 1];
  }
  return f.h;
}

// ---- B fragment: 32x16 f16, B(K,N) taken from row-major src[N][K] (i.e. stored transposed),
// stride sr halves. Same K layout as load_a: pair i -> K = 16*(i>>2) + 8*hi + 2*(i&3).
__device__ inline v16h load_bT(const _Float16* p, int sr, int n0, int k0, int lane) {
  FragH f;
  const int N  = n0 + (lane & 15);
  const int hi = lane >> 4;
  const unsigned* row = (const unsigned*)(p + N * sr);
#pragma unroll
  for (int i = 0; i < 8; ++i) {
    const int K = k0 + 16 * (i >> 2) + 8 * hi + 2 * (i & 3);
    f.u[i] = row[K >> 1];
  }
  return f.h;
}

// ---- stage a 128x128 f32 global weight into LDS as f16 transposed: sw[out*TS + in]
__device__ inline void load_wT(_Float16* sw, const float* __restrict__ Wsrc, int tid) {
  for (int idx = tid; idx < DD * DD; idx += 256) {
    const int in  = idx >> 7;
    const int out = idx & 127;
    sw[out * TS + in] = (_Float16)Wsrc[idx];
  }
}

// ---- warm a 64KB (128x128 f32) weight matrix into cache: 256 threads x 2 lines of 128B
__device__ inline void prefetch_w(const float* __restrict__ p, int tid) {
  __builtin_prefetch(p + (size_t)tid * 32, 0, 3);          // first 32KB
  __builtin_prefetch(p + 8192 + (size_t)tid * 32, 0, 3);   // second 32KB
}

// ---- 64x128 GEMM: Dst = act(A[64x128] @ W + bias). W given transposed in LDS (stride TS).
// 8 waves: wave -> rows (wave>>1)*16..+15, cols (wave&1)*64..+63 (4 WMMA col tiles).
__device__ inline void gemm_128(const _Float16* A, int as, const _Float16* W,
                                const float* __restrict__ bias, bool relu,
                                _Float16* Dst, int ds, bool dstT, int tid) {
  const int lane = tid & 31, wave = tid >> 5, hi = lane >> 4;
  const int m0 = (wave >> 1) * 16;
  const int nb = (wave & 1) * 64;
  v8f c[4];
#pragma unroll
  for (int j = 0; j < 4; ++j)
#pragma unroll
    for (int r = 0; r < 8; ++r) c[j][r] = 0.0f;

#pragma unroll
  for (int kk = 0; kk < 4; ++kk) {
    const int k0 = kk * 32;
    v16h a = load_a(A, as, m0, k0, lane);
#pragma unroll
    for (int j = 0; j < 4; ++j) {
      v16h bf = load_bT(W, TS, nb + j * 16, k0, lane);
      c[j] = __builtin_amdgcn_wmma_f32_16x16x32_f16(false, a, false, bf,
                                                    (short)0, c[j], false, false);
    }
  }
#pragma unroll
  for (int j = 0; j < 4; ++j) {
    const int col = nb + j * 16 + (lane & 15);
    const float bv = bias[col];
#pragma unroll
    for (int r = 0; r < 8; ++r) {
      float v = c[j][r] + bv;
      if (relu) v = fmaxf(v, 0.0f);
      const int row = m0 + 8 * hi + r;
      if (dstT) Dst[col * ds + row] = (_Float16)v;
      else      Dst[row * ds + col] = (_Float16)v;
    }
  }
}

// ---- final GEMM: acc += gate * exp(A @ W + bias), accumulated in registers
__device__ inline void gemm_acc(const _Float16* A, int as, const _Float16* W,
                                const float* __restrict__ bias, float gate,
                                v8f (&acc)[4], int tid) {
  const int lane = tid & 31, wave = tid >> 5;
  const int m0 = (wave >> 1) * 16;
  const int nb = (wave & 1) * 64;
  v8f c[4];
#pragma unroll
  for (int j = 0; j < 4; ++j)
#pragma unroll
    for (int r = 0; r < 8; ++r) c[j][r] = 0.0f;

#pragma unroll
  for (int kk = 0; kk < 4; ++kk) {
    const int k0 = kk * 32;
    v16h a = load_a(A, as, m0, k0, lane);
#pragma unroll
    for (int j = 0; j < 4; ++j) {
      v16h bf = load_bT(W, TS, nb + j * 16, k0, lane);
      c[j] = __builtin_amdgcn_wmma_f32_16x16x32_f16(false, a, false, bf,
                                                    (short)0, c[j], false, false);
    }
  }
#pragma unroll
  for (int j = 0; j < 4; ++j) {
    const int col = nb + j * 16 + (lane & 15);
    const float bv = bias[col];
#pragma unroll
    for (int r = 0; r < 8; ++r)
      acc[j][r] += gate * expf(c[j][r] + bv);
  }
}

// ---- attention for one head: att = softmax(q k^T / 8); o[:, head*64..] = att @ v
__device__ inline void attn_head(const _Float16* s_x, const _Float16* s_k,
                                 const _Float16* s_vT, _Float16* s_att,
                                 _Float16* s_o, int head, int tid) {
  const int lane = tid & 31, wave = tid >> 5, hi = lane >> 4;

  // --- att = q k^T * 0.125 : 64x64 output, 16 tiles, 2 per wave
#pragma unroll
  for (int t2 = 0; t2 < 2; ++t2) {
    const int idx = wave * 2 + t2;
    const int m0 = (idx >> 2) * 16, n0 = (idx & 3) * 16;
    v8f c;
#pragma unroll
    for (int r = 0; r < 8; ++r) c[r] = 0.0f;
#pragma unroll
    for (int kk = 0; kk < 2; ++kk) {
      const int k0 = head * 64 + kk * 32;
      v16h a  = load_a (s_x, TS, m0, k0, lane);
      v16h bf = load_bT(s_k, TS, n0, k0, lane);   // B(K=feat,N=t') = k[t'][feat]
      c = __builtin_amdgcn_wmma_f32_16x16x32_f16(false, a, false, bf,
                                                 (short)0, c, false, false);
    }
    const int N = n0 + (lane & 15);
#pragma unroll
    for (int r = 0; r < 8; ++r)
      s_att[(m0 + 8 * hi + r) * VS + N] = (_Float16)(c[r] * 0.125f);
  }
  __syncthreads();

  // --- row softmax (64 rows, 1 thread/row)
  if (tid < 64) {
    _Float16* rp = s_att + tid * VS;
    float m = -3.4e38f;
    for (int j = 0; j < 64; ++j) { float v = (float)rp[j]; m = fmaxf(m, v); }
    float s = 0.0f;
    for (int j = 0; j < 64; ++j) s += expf((float)rp[j] - m);
    const float inv = 1.0f / s;
    for (int j = 0; j < 64; ++j) rp[j] = (_Float16)(expf((float)rp[j] - m) * inv);
  }
  __syncthreads();

  // --- o = att @ v : 64x64, cols go to feature slice head*64..head*64+63
#pragma unroll
  for (int t2 = 0; t2 < 2; ++t2) {
    const int idx = wave * 2 + t2;
    const int m0 = (idx >> 2) * 16, n0 = (idx & 3) * 16;
    v8f c;
#pragma unroll
    for (int r = 0; r < 8; ++r) c[r] = 0.0f;
#pragma unroll
    for (int kk = 0; kk < 2; ++kk) {
      v16h a  = load_a (s_att, VS, m0, kk * 32, lane);
      v16h bf = load_bT(s_vT, VS, head * 64 + n0, kk * 32, lane); // B(K=t',N=feat)=vT[feat][t']
      c = __builtin_amdgcn_wmma_f32_16x16x32_f16(false, a, false, bf,
                                                 (short)0, c, false, false);
    }
    const int N = head * 64 + n0 + (lane & 15);
#pragma unroll
    for (int r = 0; r < 8; ++r)
      s_o[(m0 + 8 * hi + r) * TS + N] = (_Float16)c[r];
  }
  __syncthreads();
}

// ================= gating kernel: mean -> logits -> top2 softmax =================
__global__ __launch_bounds__(128)
void AMS_gate_kernel(const float* __restrict__ x, const float* __restrict__ Wg,
                     float* __restrict__ gates) {
  const int b = blockIdx.x;
  const int tid = threadIdx.x;     // 128 threads, one per feature d
  __shared__ float s_mean[DD];
  __shared__ float s_logit[EE];

  const float* xb = x + (size_t)b * TT * NN * DD;
  float s = 0.0f;
  for (int r = 0; r < TT * NN; ++r) s += xb[(size_t)r * DD + tid];
  s_mean[tid] = s * (1.0f / (TT * NN));
  __syncthreads();

  if (tid < EE) {
    float l = 0.0f;
    for (int d = 0; d < DD; ++d) l += s_mean[d] * Wg[d * EE + tid];
    s_logit[tid] = l;
  }
  __syncthreads();

  if (tid == 0) {
    int i0 = 0; float v0 = s_logit[0];
    for (int e = 1; e < EE; ++e) if (s_logit[e] > v0) { v0 = s_logit[e]; i0 = e; }
    int i1 = -1; float v1 = -3.4e38f;
    for (int e = 0; e < EE; ++e) if (e != i0 && s_logit[e] > v1) { v1 = s_logit[e]; i1 = e; }
    const float e1  = expf(v1 - v0);
    const float den = 1.0f + e1;
    for (int e = 0; e < EE; ++e) gates[b * EE + e] = 0.0f;
    gates[b * EE + i0] = 1.0f / den;
    gates[b * EE + i1] = e1 / den;
  }
}

// ================= main fused MoE kernel: one block per (b, n) =================
__global__ __launch_bounds__(256)
void AMS_moe_kernel(const float* __restrict__ x,  const float* __restrict__ Wd,
                    const float* __restrict__ bd, const float* __restrict__ Ws,
                    const float* __restrict__ bs, const float* __restrict__ gates,
                    float* __restrict__ out) {
  __shared__ _Float16 s_x [TT * TS];   // q / GEMM input tile (f16)
  __shared__ _Float16 s_w [DD * TS];   // current weight, transposed [out][in]
  __shared__ _Float16 s_k [TT * TS];   // k, later s1 (relu output)
  __shared__ _Float16 s_vT[DD * VS];   // v transposed: [feat][t]
  __shared__ _Float16 s_at[TT * VS];   // attention scores (per head, in-place softmax)
  __shared__ _Float16 s_o [TT * TS];   // attention output

  const int tid = threadIdx.x;
  const int b = blockIdx.x >> 5;       // N = 32
  const int n = blockIdx.x & 31;

  // stage x[b, :, n, :] as f16
  for (int idx = tid; idx < TT * DD; idx += 256) {
    const int t = idx >> 7, d = idx & 127;
    s_x[t * TS + d] = (_Float16)x[(((size_t)b * TT + t) * NN + n) * DD + d];
  }

  v8f acc[4];
#pragma unroll
  for (int j = 0; j < 4; ++j)
#pragma unroll
    for (int r = 0; r < 8; ++r) acc[j][r] = 0.0f;
  __syncthreads();

  for (int e = 0; e < EE; ++e) {
    const float gate = gates[b * EE + e];
    if (gate == 0.0f) continue;        // block-uniform branch (b fixed per block)

    const float* Wd0 = Wd + (((size_t)e * 2 + 0) * NN + n) * DD * DD;
    const float* Wd1 = Wd + (((size_t)e * 2 + 1) * NN + n) * DD * DD;
    const float* bd0 = bd + (((size_t)e * 2 + 0) * NN + n) * DD;
    const float* bd1 = bd + (((size_t)e * 2 + 1) * NN + n) * DD;
    const float* Ws0 = Ws + ((size_t)e * 2 + 0) * DD * DD;
    const float* Ws1 = Ws + ((size_t)e * 2 + 1) * DD * DD;
    const float* bs0 = bs + ((size_t)e * 2 + 0) * DD;
    const float* bs1 = bs + ((size_t)e * 2 + 1) * DD;

    // k = x @ Wd0 + bd0   (warm Wd1 behind the staging of Wd0)
    load_wT(s_w, Wd0, tid);
    prefetch_w(Wd1, tid);
    __syncthreads();
    gemm_128(s_x, TS, s_w, bd0, false, s_k, TS, false, tid); __syncthreads();

    // v = x @ Wd1 + bd1  (stored transposed; warm Ws0 behind it)
    load_wT(s_w, Wd1, tid);
    prefetch_w(Ws0, tid);
    __syncthreads();
    gemm_128(s_x, TS, s_w, bd1, false, s_vT, VS, true, tid); __syncthreads();

    // attention, both heads (warm Ws1 between the two heads)
    attn_head(s_x, s_k, s_vT, s_at, s_o, 0, tid);
    prefetch_w(Ws1, tid);
    attn_head(s_x, s_k, s_vT, s_at, s_o, 1, tid);

    // s1 = relu(o @ Ws0 + bs0)  (reuse s_k)
    load_wT(s_w, Ws0, tid); __syncthreads();
    gemm_128(s_o, TS, s_w, bs0, true, s_k, TS, false, tid); __syncthreads();

    // acc += gate * exp(s1 @ Ws1 + bs1)
    load_wT(s_w, Ws1, tid); __syncthreads();
    gemm_acc(s_k, TS, s_w, bs1, gate, acc, tid); __syncthreads();
  }

  // write log(combined)
  const int lane = tid & 31, wave = tid >> 5, hi = lane >> 4;
  const int m0 = (wave >> 1) * 16;
  const int nb = (wave & 1) * 64;
  const float EPSV = 2.220446049250313e-16f;
#pragma unroll
  for (int j = 0; j < 4; ++j) {
    const int col = nb + j * 16 + (lane & 15);
#pragma unroll
    for (int r = 0; r < 8; ++r) {
      const int t = m0 + 8 * hi + r;
      float v = acc[j][r];
      if (v == 0.0f) v = EPSV;
      out[(((size_t)b * TT + t) * NN + n) * DD + col] = logf(v);
    }
  }
}

extern "C" void kernel_launch(void* const* d_in, const int* in_sizes, int n_in,
                              void* d_out, int out_size, void* d_ws, size_t ws_size,
                              hipStream_t stream) {
  const float* x  = (const float*)d_in[0];
  const float* Wg = (const float*)d_in[1];
  const float* Wd = (const float*)d_in[2];
  const float* bd = (const float*)d_in[3];
  const float* Ws = (const float*)d_in[4];
  const float* bs = (const float*)d_in[5];
  float* out   = (float*)d_out;
  float* gates = (float*)d_ws;          // B*E floats

  AMS_gate_kernel<<<BB, 128, 0, stream>>>(x, Wg, gates);
  AMS_moe_kernel<<<BB * NN, 256, 0, stream>>>(x, Wd, bd, Ws, bs, gates, out);
  (void)in_sizes; (void)n_in; (void)out_size; (void)ws_size;
}
